// GNNEdgePredictor_24017457119604
// MI455X (gfx1250) — matrix operand order repaired and checked by reading, not compiled, not run on hardware
//
#include <hip/hip_runtime.h>
#include <hip/hip_bf16.h>
#include <stdint.h>

// ---------------------------------------------------------------------------
// GNN edge predictor for MI455X (gfx1250, wave32, WMMA).
// All dense math runs on v_wmma_f32_16x16x32_bf16 (bf16 inputs, f32 accum).
// Inner GEMM loop is software-pipelined: all NT B-fragments are batch-loaded
// per k-step and the next A-fragment is prefetched under the WMMA chain.
// ---------------------------------------------------------------------------

#define HID 256   // HIDDEN
#define NDIM 128  // NODE_DIM

typedef __attribute__((ext_vector_type(16))) __bf16 v16bf;
typedef __attribute__((ext_vector_type(8)))  float  v8f;

union Frag {
    v16bf   v;
    uint32_t u[8];
    uint4    q[2];
};

__device__ __forceinline__ uint16_t f2bf(float f) {
    uint32_t x = __float_as_uint(f);
    uint32_t r = x + 0x7FFFu + ((x >> 16) & 1u);   // round-to-nearest-even
    return (uint16_t)(r >> 16);
}

// A fragment: 16x32 bf16, ISA 7.12.2 layout.
// lane l: m = l&15, half = l>>4 ; vgpr v: k = (v>>2)*16 + half*8 + (v&3)*2 (+p)
// -> per lane: 16B at (k0 + half*8) and 16B at (k0 + 16 + half*8): two b128s.
__device__ __forceinline__ v16bf load_a_frag(const uint16_t* __restrict__ arow,
                                             int k0, int half) {
    Frag f;
#pragma unroll
    for (int v = 0; v < 8; ++v) {
        int k = k0 + ((v >> 2) << 4) + (half << 3) + ((v & 3) << 1);
        f.u[v] = *(const uint32_t*)(arow + k);
    }
    return f.v;
}

// B fragment pre-packed: lane's 16 bf16 values are contiguous (32B).
__device__ __forceinline__ v16bf load_b_frag(const uint32_t* __restrict__ bp) {
    Frag f;
    const uint4* q = (const uint4*)bp;
    f.q[0] = q[0];
    f.q[1] = q[1];
    return f.v;
}

// ---------------------------------------------------------------------------
// Pack f32 KxN weight matrix into WMMA B-tile layout (bf16).
// B layout (32x16 tile): lane l -> n = l&15, kgroup = l>>4 (K +8);
// within lane, element j (0..15): k = kt*32 + (j>>3)*16 + kgroup*8 + (j&7).
// Stored at P[((kt*NTtot + nt)*32 + l)*16 + j].
// ---------------------------------------------------------------------------
__global__ void __launch_bounds__(256)
pack_b_kernel(const float* __restrict__ W, uint16_t* __restrict__ P,
              int K, int Nsrc, int NTtot) {
    int t = blockIdx.x * blockDim.x + threadIdx.x;
    int total = K * NTtot * 16;
    if (t >= total) return;
    int j    = t & 15;
    int l    = (t >> 4) & 31;
    int tile = t >> 9;
    int nt   = tile % NTtot;
    int kt   = tile / NTtot;
    int k = (kt << 5) + ((j >> 3) << 4) + ((l >> 4) << 3) + (j & 7);
    int n = (nt << 4) + (l & 15);
    float val = (n < Nsrc) ? W[(size_t)k * Nsrc + n] : 0.0f;
    P[t] = f2bf(val);
}

// ---------------------------------------------------------------------------
// Tiled WMMA GEMM: each wave computes a 16 x (NT*16) strip of C = A*B (+bias).
// GATHER variant builds A rows as [node_emb[src[e]] || node_emb[dst[e]]].
// ---------------------------------------------------------------------------
template <int NT, bool GATHER, bool BIAS, bool RELU, bool BF16OUT>
__global__ void __launch_bounds__(256)
gemm_wmma_kernel(const uint16_t* __restrict__ A, int lda,
                 const int* __restrict__ src, const int* __restrict__ dst,
                 const uint16_t* __restrict__ emb,
                 const uint32_t* __restrict__ Bp, int NTtot,
                 const float* __restrict__ bias,
                 float* __restrict__ outF, uint16_t* __restrict__ outB,
                 int M, int K, int Nreal) {
    int wave = threadIdx.x >> 5;
    int lane = threadIdx.x & 31;
    int m0 = (blockIdx.x * 8 + wave) * 16;
    if (m0 >= M) return;
    int half = lane >> 4;
    int row  = m0 + (lane & 15);
    int nt0  = blockIdx.y * NT;

    const uint16_t* arow = nullptr;
    size_t sbase = 0, dbase = 0;
    if (GATHER) {
        sbase = (size_t)src[row] * HID;
        dbase = (size_t)dst[row] * HID;
    } else {
        arow = A + (size_t)row * lda;
    }

    // per-kt A fragment source (gather: first 8 steps = src row, rest = dst)
    auto fetch_a = [&](int kt) -> v16bf {
        if (GATHER) {
            const uint16_t* r = emb + ((kt < 8) ? sbase : dbase);
            return load_a_frag(r, (kt << 5) - ((kt < 8) ? 0 : HID), half);
        }
        return load_a_frag(arow, kt << 5, half);
    };

    v8f acc[NT] = {};

    const int KT = K >> 5;
    v16bf a = fetch_a(0);
    for (int kt = 0; kt < KT; ++kt) {
        // batch-load all B fragments for this k-step (one clause of b128s)
        const uint32_t* bpk = Bp + ((size_t)(kt * NTtot + nt0) * 32 + lane) * 8;
        v16bf bfr[NT];
#pragma unroll
        for (int nt = 0; nt < NT; ++nt)
            bfr[nt] = load_b_frag(bpk + (size_t)nt * 256);
        // prefetch next A fragment under the WMMA chain
        v16bf anext;
        if (kt + 1 < KT) anext = fetch_a(kt + 1);
#pragma unroll
        for (int nt = 0; nt < NT; ++nt)
            acc[nt] = __builtin_amdgcn_wmma_f32_16x16x32_bf16(
                false, a, false, bfr[nt], (short)0, acc[nt], false, false);
        a = anext;
    }

    // Epilogue: D layout -> lane holds n = lane&15; vgpr v -> m = v + 8*half.
#pragma unroll
    for (int nt = 0; nt < NT; ++nt) {
        int n = (nt0 + nt) * 16 + (lane & 15);
        if (n >= Nreal) continue;
        float bv = BIAS ? bias[n] : 0.0f;
#pragma unroll
        for (int v = 0; v < 8; ++v) {
            int m = m0 + v + (half << 3);
            float val = acc[nt][v] + bv;
            if (RELU) val = fmaxf(val, 0.0f);
            size_t off = (size_t)m * Nreal + n;
            if (BF16OUT) outB[off] = f2bf(val);
            else         outF[off] = val;
        }
    }
}

// ---------------------------------------------------------------------------
// GCN normalization + aggregation helpers
// ---------------------------------------------------------------------------
__global__ void __launch_bounds__(256)
fill_f32_kernel(float* p, float v, int n) {
    int i = blockIdx.x * blockDim.x + threadIdx.x;
    if (i < n) p[i] = v;
}

__global__ void __launch_bounds__(256)
deg_count_kernel(const int* __restrict__ dst, float* __restrict__ deg, int E) {
    int i = blockIdx.x * blockDim.x + threadIdx.x;
    if (i < E) atomicAdd(&deg[dst[i]], 1.0f);
}

__global__ void __launch_bounds__(256)
rsqrt_inplace_kernel(float* __restrict__ d, int n) {
    int i = blockIdx.x * blockDim.x + threadIdx.x;
    if (i < n) d[i] = rsqrtf(d[i] + 1.0f);
}

// out = m * dinv^2 (self loop) + bias
__global__ void __launch_bounds__(256)
init_out_kernel(const float* __restrict__ m, const float* __restrict__ dinv,
                const float* __restrict__ b, float* __restrict__ out, int N) {
    long long i = (long long)blockIdx.x * blockDim.x + threadIdx.x;
    if (i >= (long long)N * HID) return;
    int n = (int)(i >> 8);
    int c = (int)(i & (HID - 1));
    float di = dinv[n];
    out[i] = m[i] * di * di + b[c];
}

// one wave per edge: out[dst] += m[src] * dinv[src]*dinv[dst]
__global__ void __launch_bounds__(256)
edge_agg_kernel(const int* __restrict__ src, const int* __restrict__ dst,
                const float* __restrict__ m, const float* __restrict__ dinv,
                float* __restrict__ out, int E) {
    int gw = (int)(((size_t)blockIdx.x * blockDim.x + threadIdx.x) >> 5);
    int lane = threadIdx.x & 31;
    if (gw >= E) return;
    int s = src[gw], d = dst[gw];
    float w = dinv[s] * dinv[d];
    const float4* mr = (const float4*)(m + (size_t)s * HID);
    float* orow = out + (size_t)d * HID;
#pragma unroll
    for (int i = 0; i < 2; ++i) {
        float4 v = mr[lane + i * 32];
        int c = (lane + i * 32) * 4;
        atomicAdd(orow + c + 0, v.x * w);
        atomicAdd(orow + c + 1, v.y * w);
        atomicAdd(orow + c + 2, v.z * w);
        atomicAdd(orow + c + 3, v.w * w);
    }
}

__global__ void __launch_bounds__(256)
f32_to_bf16_kernel(const float* __restrict__ in, uint16_t* __restrict__ outp,
                   long long n, int relu) {
    long long i = (long long)blockIdx.x * blockDim.x + threadIdx.x;
    if (i >= n) return;
    float v = in[i];
    if (relu) v = fmaxf(v, 0.0f);
    outp[i] = f2bf(v);
}

// ---------------------------------------------------------------------------
// Launch
// ---------------------------------------------------------------------------
extern "C" void kernel_launch(void* const* d_in, const int* in_sizes, int n_in,
                              void* d_out, int out_size, void* d_ws, size_t ws_size,
                              hipStream_t stream) {
    const float* x   = (const float*)d_in[0];
    const int*   ei  = (const int*)d_in[1];
    const float* gW0 = (const float*)d_in[2];  const float* gb0 = (const float*)d_in[3];
    const float* gW1 = (const float*)d_in[4];  const float* gb1 = (const float*)d_in[5];
    const float* gW2 = (const float*)d_in[6];  const float* gb2 = (const float*)d_in[7];
    const float* cW0 = (const float*)d_in[8];  const float* cb0 = (const float*)d_in[9];
    const float* cW1 = (const float*)d_in[10]; const float* cb1 = (const float*)d_in[11];
    const float* cW2 = (const float*)d_in[12]; const float* cb2 = (const float*)d_in[13];

    const int N = in_sizes[0] / NDIM;
    const int E = in_sizes[1] / 2;
    const int* src = ei;
    const int* dst = ei + E;

    char* base = (char*)d_ws;
    size_t off = 0;
    auto alloc = [&](size_t bytes) -> void* {
        void* p = base + off;
        off += (bytes + 255) & ~(size_t)255;
        return p;
    };

    float*    dinv = (float*)   alloc((size_t)N * 4);
    uint16_t* gW0p = (uint16_t*)alloc((size_t)NDIM * HID * 2);
    uint16_t* gW1p = (uint16_t*)alloc((size_t)HID * HID * 2);
    uint16_t* gW2p = (uint16_t*)alloc((size_t)HID * HID * 2);
    uint16_t* cW0p = (uint16_t*)alloc((size_t)2 * HID * HID * 2);
    uint16_t* cW1p = (uint16_t*)alloc((size_t)HID * 128 * 2);
    uint16_t* cW2p = (uint16_t*)alloc((size_t)128 * 32 * 2);
    uint16_t* h    = (uint16_t*)alloc((size_t)N * HID * 2);
    float*    mbuf = (float*)   alloc((size_t)N * HID * 4);
    float*    obuf = (float*)   alloc((size_t)N * HID * 4);
    // e1 [E,256] bf16: reuse mbuf+obuf region if it fits, else fresh.
    uint16_t* e1;
    if ((size_t)E * HID * 2 <= (size_t)N * HID * 8) e1 = (uint16_t*)mbuf;
    else e1 = (uint16_t*)alloc((size_t)E * HID * 2);
    uint16_t* e2 = (uint16_t*)alloc((size_t)E * 128 * 2);

    auto cdiv = [](long long a, long long b) { return (int)((a + b - 1) / b); };

    // --- degree / normalization ---
    fill_f32_kernel<<<cdiv(N, 256), 256, 0, stream>>>(dinv, 0.0f, N);
    deg_count_kernel<<<cdiv(E, 256), 256, 0, stream>>>(dst, dinv, E);
    rsqrt_inplace_kernel<<<cdiv(N, 256), 256, 0, stream>>>(dinv, N);

    // --- pack weights to WMMA B layout (bf16) ---
    pack_b_kernel<<<cdiv((long long)NDIM * 16 * 16, 256), 256, 0, stream>>>(gW0, gW0p, NDIM, HID, 16);
    pack_b_kernel<<<cdiv((long long)HID  * 16 * 16, 256), 256, 0, stream>>>(gW1, gW1p, HID,  HID, 16);
    pack_b_kernel<<<cdiv((long long)HID  * 16 * 16, 256), 256, 0, stream>>>(gW2, gW2p, HID,  HID, 16);
    pack_b_kernel<<<cdiv((long long)2*HID* 16 * 16, 256), 256, 0, stream>>>(cW0, cW0p, 2*HID, HID, 16);
    pack_b_kernel<<<cdiv((long long)HID  *  8 * 16, 256), 256, 0, stream>>>(cW1, cW1p, HID,  128, 8);
    pack_b_kernel<<<cdiv((long long)128  *  2 * 16, 256), 256, 0, stream>>>(cW2, cW2p, 128,   18, 2);

    // --- x -> bf16 ---
    f32_to_bf16_kernel<<<cdiv((long long)N * NDIM, 256), 256, 0, stream>>>(x, h, (long long)N * NDIM, 0);

    // --- 3 GCN layers ---
    const uint16_t* gWp[3] = {gW0p, gW1p, gW2p};
    const float*    gbv[3] = {gb0, gb1, gb2};
    int Kl[3] = {NDIM, HID, HID};
    for (int i = 0; i < 3; ++i) {
        dim3 g(cdiv(N, 128), 2);
        gemm_wmma_kernel<8, false, false, false, false><<<g, 256, 0, stream>>>(
            h, Kl[i], nullptr, nullptr, nullptr,
            (const uint32_t*)gWp[i], 16, nullptr, mbuf, nullptr, N, Kl[i], HID);
        init_out_kernel<<<cdiv((long long)N * HID, 256), 256, 0, stream>>>(mbuf, dinv, gbv[i], obuf, N);
        edge_agg_kernel<<<cdiv((long long)E * 32, 256), 256, 0, stream>>>(src, dst, mbuf, dinv, obuf, E);
        f32_to_bf16_kernel<<<cdiv((long long)N * HID, 256), 256, 0, stream>>>(
            obuf, h, (long long)N * HID, (i < 2) ? 1 : 0);
    }

    // --- edge classifier MLP ---
    {   // layer 0: gather [src||dst] -> [E,512] @ cW0 -> relu -> bf16 [E,256]
        dim3 g(cdiv(E, 128), 2);
        gemm_wmma_kernel<8, true, true, true, true><<<g, 256, 0, stream>>>(
            nullptr, 0, src, dst, h,
            (const uint32_t*)cW0p, 16, cb0, nullptr, e1, E, 2 * HID, HID);
    }
    {   // layer 1: [E,256] @ cW1 -> relu -> bf16 [E,128]
        dim3 g(cdiv(E, 128), 1);
        gemm_wmma_kernel<8, false, true, true, true><<<g, 256, 0, stream>>>(
            e1, HID, nullptr, nullptr, nullptr,
            (const uint32_t*)cW1p, 8, cb1, nullptr, e2, E, HID, 128);
    }
    {   // layer 2: [E,128] @ cW2 (padded N=32) -> f32 logits [E,18]
        dim3 g(cdiv(E, 128), 1);
        gemm_wmma_kernel<2, false, true, false, false><<<g, 256, 0, stream>>>(
            e2, 128, nullptr, nullptr, nullptr,
            (const uint32_t*)cW2p, 2, cb2, (float*)d_out, nullptr, E, 128, 18);
    }
}
